// SpectralConvBlock_40321152974862
// MI455X (gfx1250) — matrix operand order repaired and checked by reading, not compiled, not run on hardware
//
#include <hip/hip_runtime.h>
#include <hip/hip_bf16.h>

typedef __attribute__((ext_vector_type(16))) _Float16 v16h;
typedef __attribute__((ext_vector_type(8)))  float    v8f;
typedef __attribute__((ext_vector_type(4)))  int      v4i;
typedef __attribute__((ext_vector_type(2)))  int      v2i;

#define CIN  64
#define COUT 128
#define HH   512
#define WW   512
#define KTOT 576            // CIN * 9, K reordered as (kh*3+kw)*64 + ci
#define NSTEP 18            // KTOT / 32
#define HWN  (HH * WW)
#define CIPAD 80            // 64 ci + 16 pad (keeps 32B alignment, spreads banks)

// CDNA5 async global->LDS path (ASYNCcnt), guarded so both toolchains compile.
#if __has_builtin(__builtin_amdgcn_global_load_async_to_lds_b128) && \
    __has_builtin(__builtin_amdgcn_global_load_async_to_lds_b64)  && \
    __has_builtin(__builtin_amdgcn_s_wait_asynccnt)
#define USE_ASYNC_LDS 1
#endif

#define ASG __attribute__((address_space(1)))
#define ASL __attribute__((address_space(3)))

// ---------------------------------------------------------------------------
// Conv 3x3 as implicit GEMM with V_WMMA_F32_16X16X32_F16.
// Block: 128 threads = 4 waves. Each wave: one 16(co) x 16(px) tile.
// K ordering: k = (kh*3+kw)*64 + ci  -> B fragments are contiguous ci runs.
// A is staged in fragment order so each step is one aligned v16h LDS load.
// ---------------------------------------------------------------------------
__global__ __launch_bounds__(128) void conv_wmma_kernel(
    const float* __restrict__ x,      // (CIN, H, W)
    const float* __restrict__ wgt,    // (COUT, CIN, 3, 3)
    const float* __restrict__ bias,   // (COUT)
    float2* __restrict__ yc)          // (COUT, H, W) complex
{
    __shared__ __attribute__((aligned(32))) _Float16 A_frag[NSTEP * 32 * 16]; // 18 KB
    __shared__ __attribute__((aligned(32))) _Float16 Xs[3 * 66 * CIPAD];      // 31 KB

    const int tid  = threadIdx.x;
    const int lane = tid & 31;
    const int wave = tid >> 5;

    const int co_base = blockIdx.z * 16;
    const int h       = blockIdx.y;
    const int wblk    = blockIdx.x * 64;

    // --- stage A in fragment order, contiguous k = 16*hi + i per 32-k step ---
    for (int idx = tid; idx < NSTEP * 32 * 16; idx += 128) {
        int s      = idx >> 9;
        int lane_l = (idx >> 4) & 31;
        int i      = idx & 15;
        int m  = lane_l & 15;
        int hi = lane_l >> 4;
        int p  = i >> 1, o = i & 1;
        int ka = s * 32 + (hi << 4) + i;
        int ci = ka & 63;          // reordered K: k = r*64 + ci
        int r  = ka >> 6;          // r = kh*3 + kw
        A_frag[idx] = (_Float16)wgt[(co_base + m) * KTOT + ci * 9 + r];
    }

    // --- stage x slab: [kh][ww][ci] (ci contiguous, padded to CIPAD) ---
    for (int i = tid; i < CIN * 3 * 66; i += 128) {
        int ci = i / (3 * 66);
        int r  = i % (3 * 66);
        int kh = r / 66;
        int ww = r % 66;
        int gh = h + kh - 1;
        int gw = wblk + ww - 1;
        float v = 0.0f;
        if (gh >= 0 && gh < HH && gw >= 0 && gw < WW)
            v = x[(size_t)ci * HWN + gh * WW + gw];
        Xs[(kh * 66 + ww) * CIPAD + ci] = (_Float16)v;
    }
    __syncthreads();

    const int hi    = lane >> 4;          // half-wave: K sub-range / D row-high
    const int ncol  = lane & 15;          // B / D column
    const int wloc  = wave * 16 + ncol;   // pixel offset within 64-px block
    const int khalf = hi << 4;

    v8f acc = {};

#pragma unroll
    for (int s = 0; s < NSTEP; ++s) {
        v16h a = *(const v16h*)&A_frag[(s * 32 + lane) * 16];
        int kk  = s * 32 + khalf;         // reordered-K base for this half
        int r   = kk >> 6;                // kh*3 + kw
        int ci0 = kk & 63;                // multiple of 16
        int kh  = r / 3;
        int kw  = r - kh * 3;
        v16h b = *(const v16h*)&Xs[(kh * 66 + wloc + kw) * CIPAD + ci0];
        acc = __builtin_amdgcn_wmma_f32_16x16x32_f16(
            /*neg_a=*/false, a, /*neg_b=*/false, b,
            /*c_mod=*/(short)0, acc, /*reuse_a=*/false, /*reuse_b=*/false);
    }

    // D layout: VGPR r -> M = r + hi*8, col = ncol
    const int n = h * WW + wblk + wloc;
#pragma unroll
    for (int r = 0; r < 8; ++r) {
        int co = co_base + r + (hi << 3);
        float v = acc[r] + bias[co];
        yc[(size_t)co * HWN + n] = make_float2(v, 0.0f);
    }
}

// ---------------------------------------------------------------------------
// 512-point Stockham radix-2 FFT in LDS. 256 threads, self-sorting, ping-pong.
// sign = -1 forward, +1 inverse (unnormalized). Returns buffer holding result.
// ---------------------------------------------------------------------------
__device__ inline float2 cmul(float2 a, float2 b) {
    return make_float2(a.x * b.x - a.y * b.y, a.x * b.y + a.y * b.x);
}

__device__ float2* fft512(float2* bufA, float2* bufB, int t, float sign) {
    float2* src = bufA;
    float2* dst = bufB;
#pragma unroll
    for (int ns = 1; ns < 512; ns <<= 1) {
        __syncthreads();
        int k = t & (ns - 1);
        float ang = sign * 3.14159265358979323846f * (float)k / (float)ns;
        float c = __cosf(ang);
        float s = __sinf(ang);
        float2 u  = src[t];
        float2 v  = src[t + 256];
        float2 tv = cmul(make_float2(c, s), v);
        int idx = ((t - k) << 1) + k;          // (t/ns)*2ns + k
        dst[idx]      = make_float2(u.x + tv.x, u.y + tv.y);
        dst[idx + ns] = make_float2(u.x - tv.x, u.y - tv.y);
        float2* tmp = src; src = dst; dst = tmp;
    }
    __syncthreads();
    return src;
}

// Load one contiguous 512-complex row into LDS (async DMA path when present).
__device__ inline void load_row_to_lds(const float2* g, float2* buf, int t) {
#ifdef USE_ASYNC_LDS
    // 256 threads x 16B = 4KB: elements 2t, 2t+1 per thread
    __builtin_amdgcn_global_load_async_to_lds_b128(
        (ASG v4i*)(g + 2 * t), (ASL v4i*)(buf + 2 * t), 0, 0);
    __builtin_amdgcn_s_wait_asynccnt(0);
#else
    buf[t]       = g[t];
    buf[t + 256] = g[t + 256];
#endif
}

// Load one strided column (stride WW complex) into LDS.
__device__ inline void load_col_to_lds(const float2* g, float2* buf, int t) {
#ifdef USE_ASYNC_LDS
    __builtin_amdgcn_global_load_async_to_lds_b64(
        (ASG v2i*)(g + (size_t)t * WW), (ASL v2i*)(buf + t), 0, 0);
    __builtin_amdgcn_global_load_async_to_lds_b64(
        (ASG v2i*)(g + (size_t)(t + 256) * WW), (ASL v2i*)(buf + t + 256), 0, 0);
    __builtin_amdgcn_s_wait_asynccnt(0);
#else
    buf[t]       = g[(size_t)t * WW];
    buf[t + 256] = g[(size_t)(t + 256) * WW];
#endif
}

// Forward FFT over rows (contiguous, coalesced), in place.
__global__ __launch_bounds__(256) void fft_rows_fwd(float2* __restrict__ yc)
{
    __shared__ float2 bufA[512], bufB[512];
    const int t = threadIdx.x;
    const size_t base = ((size_t)blockIdx.y * HH + blockIdx.x) * WW; // (c, h)
    load_row_to_lds(yc + base, bufA, t);
    float2* res = fft512(bufA, bufB, t, -1.0f);
    yc[base + t]       = res[t];
    yc[base + t + 256] = res[t + 256];
}

// Fused column pass: forward FFT, mask multiply, inverse FFT. In place.
__global__ __launch_bounds__(256) void fft_cols_fused(
    float2* __restrict__ yc, const float* __restrict__ mask)
{
    __shared__ float2 bufA[512], bufB[512];
    const int t = threadIdx.x;
    const int c = blockIdx.y;
    const int x = blockIdx.x;
    const size_t base = (size_t)c * HWN + x;
    load_col_to_lds(yc + base, bufA, t);

    float2* res = fft512(bufA, bufB, t, -1.0f);

    float m0 = mask[((size_t)c * HH + t) * WW + x];
    float m1 = mask[((size_t)c * HH + t + 256) * WW + x];
    res[t].x       *= m0; res[t].y       *= m0;
    res[t + 256].x *= m1; res[t + 256].y *= m1;

    float2* other = (res == bufA) ? bufB : bufA;
    float2* res2  = fft512(res, other, t, +1.0f);

    yc[base + (size_t)t * WW]         = res2[t];
    yc[base + (size_t)(t + 256) * WW] = res2[t + 256];
}

// Inverse FFT over rows, write real part * 1/(H*W).
__global__ __launch_bounds__(256) void ifft_rows_final(
    const float2* __restrict__ yc, float* __restrict__ out)
{
    __shared__ float2 bufA[512], bufB[512];
    const int t = threadIdx.x;
    const size_t base = ((size_t)blockIdx.y * HH + blockIdx.x) * WW; // (c, h)
    load_row_to_lds(yc + base, bufA, t);
    float2* res = fft512(bufA, bufB, t, +1.0f);
    const float scale = 1.0f / ((float)HH * (float)WW);
    out[base + t]       = res[t].x * scale;
    out[base + t + 256] = res[t + 256].x * scale;
}

// ---------------------------------------------------------------------------
extern "C" void kernel_launch(void* const* d_in, const int* in_sizes, int n_in,
                              void* d_out, int out_size, void* d_ws, size_t ws_size,
                              hipStream_t stream) {
    const float* x        = (const float*)d_in[0];
    const float* conv_w   = (const float*)d_in[1];
    const float* conv_b   = (const float*)d_in[2];
    const float* fft_mask = (const float*)d_in[3];
    float*  out = (float*)d_out;
    float2* yc  = (float2*)d_ws;   // (COUT, H, W) complex, 256 MB

    // Conv: grid = (64-px groups per row, rows, co-tiles)
    dim3 gconv(WW / 64, HH, COUT / 16);
    conv_wmma_kernel<<<gconv, 128, 0, stream>>>(x, conv_w, conv_b, yc);

    dim3 gfft(WW, COUT);  // (h|x, channel)
    fft_rows_fwd   <<<gfft, 256, 0, stream>>>(yc);
    fft_cols_fused <<<gfft, 256, 0, stream>>>(yc, fft_mask);
    ifft_rows_final<<<gfft, 256, 0, stream>>>(yc, out);
}
